// ClusteringLayer_12154757447929
// MI455X (gfx1250) — matrix operand order repaired and checked by reading, not compiled, not run on hardware
//
#include <hip/hip_runtime.h>

typedef __attribute__((ext_vector_type(16))) __bf16          v16bf;
typedef __attribute__((ext_vector_type(16))) unsigned short  v16u;
typedef __attribute__((ext_vector_type(8)))  float           v8f;
typedef __attribute__((ext_vector_type(4)))  float           v4f;

#define N_ROWS  131072
#define DIM     256
#define NCLUST  256
#define TILES   16          // 256 clusters / 16
#define KSTEPS  8           // 256 dim / 32
#define LDSPITCH 40         // ushorts per staged w row: 80B = 20 banks -> conflict free, 16B aligned

__device__ __forceinline__ unsigned short f32_to_bf16_rne(float f) {
    unsigned u = __builtin_bit_cast(unsigned, f);
    unsigned r = u + 0x7FFFu + ((u >> 16) & 1u);
    return (unsigned short)(r >> 16);
}
__device__ __forceinline__ float bf16_to_f32(unsigned short h) {
    return __builtin_bit_cast(float, (unsigned)h << 16);
}

// ---------------------------------------------------------------------------
// Prep: split w into bf16 hi/lo planes, compute ||w_k||^2 per cluster.
// grid = 256 blocks (one per cluster), 256 threads.
// ---------------------------------------------------------------------------
__global__ void cluster_prep(const float* __restrict__ w,
                             unsigned short* __restrict__ w_hi,
                             unsigned short* __restrict__ w_lo,
                             float* __restrict__ nb) {
    const int c = blockIdx.x;
    const int t = threadIdx.x;
    float x = w[(size_t)c * DIM + t];
    unsigned short h = f32_to_bf16_rne(x);
    w_hi[(size_t)c * DIM + t] = h;
    w_lo[(size_t)c * DIM + t] = f32_to_bf16_rne(x - bf16_to_f32(h));

    float sq = x * x;
    #pragma unroll
    for (int off = 1; off < 32; off <<= 1) sq += __shfl_xor(sq, off, 32);
    __shared__ float part[8];
    if ((t & 31) == 0) part[t >> 5] = sq;
    __syncthreads();
    if (t == 0) {
        float s = 0.f;
        #pragma unroll
        for (int i = 0; i < 8; ++i) s += part[i];
        nb[c] = s;
    }
}

// ---------------------------------------------------------------------------
// Main: each wave computes 16 rows x 256 clusters with bf16x3 split WMMA.
// Block = 256 threads = 8 waves = 128 rows. Grid = 131072/128 = 1024 blocks.
// ---------------------------------------------------------------------------
__global__ __launch_bounds__(256, 1)
void cluster_main(const float* __restrict__ Z,
                  const unsigned short* __restrict__ w_hi,
                  const unsigned short* __restrict__ w_lo,
                  const float* __restrict__ nb,
                  float* __restrict__ out) {
    __shared__ unsigned short sWhi[NCLUST * LDSPITCH];   // 20 KB
    __shared__ unsigned short sWlo[NCLUST * LDSPITCH];   // 20 KB

    const int lane    = threadIdx.x & 31;
    const int wave    = threadIdx.x >> 5;
    const int rowbase = blockIdx.x * 128 + wave * 16;
    const int mrow    = lane & 15;            // A-fragment row
    const bool hihalf = lane >= 16;
    const int sel_a   = hihalf ? 16 : 0;      // A K-offset (contiguous k = 16h + s)
    const int sel_b   = hihalf ? 16 : 0;      // B K-offset (contiguous k = 16h + s)
    const int ncol    = lane & 15;            // B/C/D column within tile

    const float* zrow = Z + (size_t)(rowbase + mrow) * DIM;

    v8f acc[TILES];
    #pragma unroll
    for (int t = 0; t < TILES; ++t) acc[t] = (v8f){0.f,0.f,0.f,0.f,0.f,0.f,0.f,0.f};

    float na = 0.f;

    for (int ks = 0; ks < KSTEPS; ++ks) {
        const int kb = ks * 32;

        // ---- stage this K-slab of w_hi/w_lo into LDS (256 rows x 32 bf16) ----
        __syncthreads();
        {
            const int c = threadIdx.x;                       // cluster row
            const uint4* gh = (const uint4*)(w_hi + (size_t)c * DIM + kb);
            const uint4* gl = (const uint4*)(w_lo + (size_t)c * DIM + kb);
            uint4* dh = (uint4*)(&sWhi[c * LDSPITCH]);
            uint4* dl = (uint4*)(&sWlo[c * LDSPITCH]);
            #pragma unroll
            for (int i = 0; i < 4; ++i) { dh[i] = gh[i]; dl[i] = gl[i]; }
        }
        __syncthreads();

        // ---- build A hi/lo fragments (16x32 bf16, contiguous k = 16h + s) ----
        v4f f0 = *(const v4f*)(zrow + kb + sel_a);
        v4f f1 = *(const v4f*)(zrow + kb + sel_a + 4);
        v4f f2 = *(const v4f*)(zrow + kb + sel_a + 8);
        v4f f3 = *(const v4f*)(zrow + kb + sel_a + 12);
        float fv[16];
        #pragma unroll
        for (int i = 0; i < 4; ++i) { fv[i]=f0[i]; fv[4+i]=f1[i]; fv[8+i]=f2[i]; fv[12+i]=f3[i]; }

        v16u ahu, alu;
        #pragma unroll
        for (int i = 0; i < 16; ++i) {
            float x = fv[i];
            na += x * x;
            unsigned short h = f32_to_bf16_rne(x);
            ahu[i] = h;
            alu[i] = f32_to_bf16_rne(x - bf16_to_f32(h));
        }
        v16bf a_hi = __builtin_bit_cast(v16bf, ahu);
        v16bf a_lo = __builtin_bit_cast(v16bf, alu);

        // ---- 16 column tiles: hi*hi + hi*lo + lo*hi ----
        #pragma unroll
        for (int t = 0; t < TILES; ++t) {
            const int crow = t * 16 + ncol;
            v16bf b_hi = *(const v16bf*)(&sWhi[crow * LDSPITCH + sel_b]);
            v16bf b_lo = *(const v16bf*)(&sWlo[crow * LDSPITCH + sel_b]);
            acc[t] = __builtin_amdgcn_wmma_f32_16x16x32_bf16(false, a_hi, false, b_hi,
                                                             (short)0, acc[t], false, false);
            acc[t] = __builtin_amdgcn_wmma_f32_16x16x32_bf16(false, a_hi, false, b_lo,
                                                             (short)0, acc[t], false, false);
            acc[t] = __builtin_amdgcn_wmma_f32_16x16x32_bf16(false, a_lo, false, b_hi,
                                                             (short)0, acc[t], false, false);
        }
    }

    // ---- complete ||z||^2 (two lanes cover disjoint halves of each row) ----
    na += __shfl_xor(na, 16, 32);
    float na_r[8];
    #pragma unroll
    for (int r = 0; r < 8; ++r)
        na_r[r] = __shfl(na, r + (hihalf ? 8 : 0), 32);   // C row = r + 8*half

    // ---- fused epilogue: q = 1/(1 + max(na + 2ab - nb, 0)), in place ----
    #pragma unroll
    for (int t = 0; t < TILES; ++t) {
        float nbv = nb[t * 16 + ncol];
        #pragma unroll
        for (int r = 0; r < 8; ++r) {
            float d2 = fmaxf(na_r[r] + 2.f * acc[t][r] - nbv, 0.f);
            acc[t][r] = 1.f / (1.f + d2);
        }
    }

    // ---- row sums across the 16 lanes of each half + all tiles ----
    float rinv[8];
    #pragma unroll
    for (int r = 0; r < 8; ++r) {
        float s = 0.f;
        #pragma unroll
        for (int t = 0; t < TILES; ++t) s += acc[t][r];
        s += __shfl_xor(s, 1, 32);
        s += __shfl_xor(s, 2, 32);
        s += __shfl_xor(s, 4, 32);
        s += __shfl_xor(s, 8, 32);
        rinv[r] = 1.f / s;
    }

    // ---- store normalized q (coalesced 64B per half-wave row segment) ----
    #pragma unroll
    for (int r = 0; r < 8; ++r) {
        float* orow = out + (size_t)(rowbase + r + (hihalf ? 8 : 0)) * NCLUST;
        #pragma unroll
        for (int t = 0; t < TILES; ++t)
            orow[t * 16 + ncol] = acc[t][r] * rinv[r];
    }
}

// ---------------------------------------------------------------------------
extern "C" void kernel_launch(void* const* d_in, const int* in_sizes, int n_in,
                              void* d_out, int out_size, void* d_ws, size_t ws_size,
                              hipStream_t stream) {
    const float* Z = (const float*)d_in[0];
    const float* w = (const float*)d_in[1];
    float* out = (float*)d_out;

    unsigned short* w_hi = (unsigned short*)d_ws;                 // 128 KB
    unsigned short* w_lo = w_hi + (size_t)NCLUST * DIM;           // 128 KB
    float*          nb   = (float*)(w_lo + (size_t)NCLUST * DIM); // 1 KB

    cluster_prep<<<NCLUST, 256, 0, stream>>>(w, w_hi, w_lo, nb);
    cluster_main<<<N_ROWS / 128, 256, 0, stream>>>(Z, w_hi, w_lo, nb, out);
}
